// EncoderDecoder_75007308857366
// MI455X (gfx1250) — compile-verified, matrix operand order checked by reading, not run on hardware
//
#include <hip/hip_runtime.h>

// ---------------- problem constants (match reference) ----------------
#define IN_DIM   32000
#define OUT_DIM  32000
#define HID      1024
#define MAXLEN   32
#define BSZ      64
#define SRC_LEN  256

// ---------------- WMMA vector types (gfx1250, wave32) ----------------
typedef __attribute__((ext_vector_type(16))) __bf16 bf16x16;
typedef __attribute__((ext_vector_type(8)))  __bf16 bf16x8;
typedef __attribute__((ext_vector_type(8)))  float  f32x8;

__device__ __forceinline__ float sigmoidf_(float x) {
    return 1.0f / (1.0f + __expf(-x));
}

// =====================================================================
// Transpose + fp32->bf16 weight prep:  dst[c*ldd + coloff + r] = src[r*C + c]
// =====================================================================
__global__ void transpose_bf16_kernel(const float* __restrict__ src,
                                      __bf16* __restrict__ dst,
                                      int R, int C, int ldd, int coloff)
{
    size_t i    = (size_t)blockIdx.x * blockDim.x + threadIdx.x;
    size_t tot  = (size_t)R * C;
    size_t step = (size_t)gridDim.x * blockDim.x;
    for (; i < tot; i += step) {
        size_t r = i / C, c = i % C;
        dst[c * (size_t)ldd + coloff + r] = (__bf16)src[i];
    }
}

// =====================================================================
// bf16 WMMA GEMM, register-blocked + software double-buffered:
//   each wave computes a 16 x (16*NB) strip of C = A[M,K] * BT[N,K]^T.
//   A fragment reused across NB B tiles; k-loop unrolled x2 with two
//   fragment buffer sets so loads for k+32 overlap WMMAs on k.
//   Requires K % 64 == 0 (true for all GEMMs here: K in {1024, 2048}).
// Fragment layout (16-bit operands, wave32):
//   lane l: hb = l>>4, lr = l&15
//   elements 0..7  <- K = k0 + 8*hb      .. +7   (one b128 load)
//   elements 8..15 <- K = k0 + 16 + 8*hb .. +7   (one b128 load)
// D layout: vgpr r -> (m = 16*mt + r + 8*hb, n = 16*nt + lr)
// =====================================================================
template <int NB>
__global__ __launch_bounds__(256)
void gemm_bf16_wmma(const __bf16* __restrict__ A,
                    const __bf16* __restrict__ BT,
                    const float*  __restrict__ bias,
                    float*        __restrict__ C,
                    __bf16*       __restrict__ Cb,
                    int M, int N, int K, int act)
{
    const int lane   = threadIdx.x & 31;
    const int hb     = lane >> 4;
    const int lr     = lane & 15;
    const int wave   = (int)((blockIdx.x * (size_t)blockDim.x + threadIdx.x) >> 5);
    const int nwav   = (int)((gridDim.x * (size_t)blockDim.x) >> 5);
    const int mt_n   = M >> 4;
    const int ns_n   = N / (16 * NB);
    const int nstrip = mt_n * ns_n;

    union F { bf16x16 v; bf16x8 h[2]; };

    for (int sidx = wave; sidx < nstrip; sidx += nwav) {
        const int mt = sidx / ns_n;
        const int ns = sidx - mt * ns_n;

        const __bf16* arow = A + (size_t)(mt * 16 + lr) * K + hb * 8;
        const __bf16* brow[NB];
#pragma unroll
        for (int u = 0; u < NB; ++u)
            brow[u] = BT + (size_t)(ns * 16 * NB + u * 16 + lr) * K + hb * 8;

        f32x8 acc[NB];
#pragma unroll
        for (int u = 0; u < NB; ++u)
            acc[u] = (f32x8){0.f, 0.f, 0.f, 0.f, 0.f, 0.f, 0.f, 0.f};

        // ---- preload k-slice 0 into buffer 0 ----
        F a0, a1, b0[NB], b1[NB];
        a0.h[0] = *(const bf16x8*)(arow + 0);
        a0.h[1] = *(const bf16x8*)(arow + 16);
#pragma unroll
        for (int u = 0; u < NB; ++u) {
            b0[u].h[0] = *(const bf16x8*)(brow[u] + 0);
            b0[u].h[1] = *(const bf16x8*)(brow[u] + 16);
        }

        for (int k0 = 0; k0 < K; k0 += 64) {
            // load k0+32 slice into buffer 1 (always valid: K % 64 == 0)
            a1.h[0] = *(const bf16x8*)(arow + k0 + 32);
            a1.h[1] = *(const bf16x8*)(arow + k0 + 48);
#pragma unroll
            for (int u = 0; u < NB; ++u) {
                b1[u].h[0] = *(const bf16x8*)(brow[u] + k0 + 32);
                b1[u].h[1] = *(const bf16x8*)(brow[u] + k0 + 48);
                __builtin_prefetch(brow[u] + k0 + 288, 0, 0);   // global_prefetch_b8 -> GL2
            }
            // WMMAs on buffer 0 overlap buffer-1 loads
#pragma unroll
            for (int u = 0; u < NB; ++u)
                acc[u] = __builtin_amdgcn_wmma_f32_16x16x32_bf16(
                             false, a0.v, false, b0[u].v, (short)0, acc[u],
                             false, false);

            if (k0 + 64 < K) {          // load next iteration's buffer 0
                a0.h[0] = *(const bf16x8*)(arow + k0 + 64);
                a0.h[1] = *(const bf16x8*)(arow + k0 + 80);
#pragma unroll
                for (int u = 0; u < NB; ++u) {
                    b0[u].h[0] = *(const bf16x8*)(brow[u] + k0 + 64);
                    b0[u].h[1] = *(const bf16x8*)(brow[u] + k0 + 80);
                }
            }
            // WMMAs on buffer 1 overlap next buffer-0 loads
#pragma unroll
            for (int u = 0; u < NB; ++u)
                acc[u] = __builtin_amdgcn_wmma_f32_16x16x32_bf16(
                             false, a1.v, false, b1[u].v, (short)0, acc[u],
                             false, false);
        }

#pragma unroll
        for (int u = 0; u < NB; ++u) {
            const int n  = ns * 16 * NB + u * 16 + lr;
            const float bv = bias ? bias[n] : 0.f;
#pragma unroll
            for (int r = 0; r < 8; ++r) {
                const int m = mt * 16 + hb * 8 + r;
                float v = acc[u][r] + bv;
                if (act == 1) v = tanhf(v);
                if (C)  C [(size_t)m * N + n] = v;
                if (Cb) Cb[(size_t)m * N + n] = (__bf16)v;
            }
        }
    }
}

// =====================================================================
// Encoder init: h=c=0, A = [bf16(enc_emb[source[:,0]]), bf16(0)]
// =====================================================================
__global__ void enc_init_kernel(const int* __restrict__ source,
                                const float* __restrict__ emb,
                                float* __restrict__ h, float* __restrict__ c,
                                __bf16* __restrict__ Abuf)
{
    int idx = blockIdx.x * blockDim.x + threadIdx.x;
    if (idx >= BSZ * HID) return;
    int b = idx / HID, j = idx % HID;
    h[idx] = 0.f; c[idx] = 0.f;
    Abuf[(size_t)b * 2 * HID + HID + j] = (__bf16)0.f;
    int w = source[b * SRC_LEN];
    Abuf[(size_t)b * 2 * HID + j] = (__bf16)emb[(size_t)w * HID + j];
}

// =====================================================================
// LSTM cell elementwise (+ optional encoder mask, hs dump, next-x gather)
// z layout: [B][4*HID] = i | f | g | o   (bias already fused by GEMM)
// =====================================================================
__global__ void lstm_cell_kernel(const float* __restrict__ z,
                                 float* __restrict__ h, float* __restrict__ c,
                                 const int* __restrict__ source, int t,
                                 float* __restrict__ hs, __bf16* __restrict__ hs_bf,
                                 __bf16* __restrict__ Abuf,
                                 const float* __restrict__ emb)
{
    int idx = blockIdx.x * blockDim.x + threadIdx.x;
    if (idx >= BSZ * HID) return;
    int b = idx / HID, j = idx % HID;
    const float* zr = z + (size_t)b * 4 * HID;
    float iv = sigmoidf_(zr[j]);
    float fv = sigmoidf_(zr[HID + j]);
    float gv = tanhf(zr[2 * HID + j]);
    float ov = sigmoidf_(zr[3 * HID + j]);
    float cn = fv * c[idx] + iv * gv;
    float hn = ov * tanhf(cn);

    bool msk = true;
    if (source) msk = source[b * SRC_LEN + t] > 0;
    float hv = msk ? hn : h[idx];
    float cv = msk ? cn : c[idx];
    h[idx] = hv; c[idx] = cv;

    if (hs) {
        size_t o = ((size_t)b * SRC_LEN + t) * HID + j;
        hs[o]    = hv;
        hs_bf[o] = (__bf16)hv;
    }
    Abuf[(size_t)b * 2 * HID + HID + j] = (__bf16)hv;     // h-half for next step
    if (source && (t + 1) < SRC_LEN) {                    // next x gather (enc)
        int w = source[b * SRC_LEN + t + 1];
        Abuf[(size_t)b * 2 * HID + j] = (__bf16)emb[(size_t)w * HID + j];
    }
}

// =====================================================================
// Decoder embedding gather (y0 == 1)
// =====================================================================
__global__ void dec_embed_kernel(const float* __restrict__ emb,
                                 const int* __restrict__ y, int t,
                                 __bf16* __restrict__ Abuf)
{
    int idx = blockIdx.x * blockDim.x + threadIdx.x;
    if (idx >= BSZ * HID) return;
    int b = idx / HID, j = idx % HID;
    int w = (t == 0) ? 1 : y[b];
    Abuf[(size_t)b * 2 * HID + j] = (__bf16)emb[(size_t)w * HID + j];
}

// =====================================================================
// Attention: scores = ht . hsW[k], masked softmax (max over ALL keys,
// as in reference), ctx = sum a_k hs[k]; writes cat = [bf16(ctx), bf16(ht)]
// One block (256 threads) per batch row; 256 keys = 1 key / thread.
// =====================================================================
__global__ __launch_bounds__(256)
void attention_kernel(const float* __restrict__ h,
                      const __bf16* __restrict__ hsW,
                      const float* __restrict__ hs,
                      const int* __restrict__ source,
                      __bf16* __restrict__ cat)
{
    __shared__ float sm[SRC_LEN];
    __shared__ float red[256];
    const int b = blockIdx.x, tid = threadIdx.x;

    const bf16x8* wv   = (const bf16x8*)(hsW + ((size_t)b * SRC_LEN + tid) * HID);
    const float*  hrow = h + (size_t)b * HID;
    float s = 0.f;
    for (int j = 0; j < HID / 8; ++j) {
        bf16x8 w8 = wv[j];
        const float* hp = hrow + j * 8;
#pragma unroll
        for (int e = 0; e < 8; ++e) s += hp[e] * (float)w8[e];
    }

    red[tid] = s;
    __syncthreads();
    for (int st = 128; st > 0; st >>= 1) {
        if (tid < st) red[tid] = fmaxf(red[tid], red[tid + st]);
        __syncthreads();
    }
    float mx = red[0];
    __syncthreads();

    float m = (source[b * SRC_LEN + tid] > 0) ? 1.f : 0.f;
    float e = __expf(s - mx) * m;
    sm[tid]  = e;
    red[tid] = e;
    __syncthreads();
    for (int st = 128; st > 0; st >>= 1) {
        if (tid < st) red[tid] += red[tid + st];
        __syncthreads();
    }
    float inv = 1.f / red[0];
    __syncthreads();

    for (int j = tid; j < HID; j += 256) {
        float acc = 0.f;
        for (int k = 0; k < SRC_LEN; ++k)
            acc += sm[k] * hs[((size_t)b * SRC_LEN + k) * HID + j];
        cat[(size_t)b * 2 * HID + j]       = (__bf16)(acc * inv);
        cat[(size_t)b * 2 * HID + HID + j] = (__bf16)hrow[j];
    }
}

// =====================================================================
// Row softmax over 32000 logits -> d_out[b,t,:], first-occurrence argmax -> y
// =====================================================================
__global__ __launch_bounds__(256)
void softmax_argmax_kernel(const float* __restrict__ logits, int t,
                           float* __restrict__ out, int* __restrict__ y)
{
    __shared__ float rv[256];
    __shared__ int   ri[256];
    const int b = blockIdx.x, tid = threadIdx.x;
    const float* row = logits + (size_t)b * OUT_DIM;

    float mx = -3.4e38f; int mi = 0x7fffffff;
    for (int v = tid; v < OUT_DIM; v += 256) {
        float x = row[v];
        if (x > mx) { mx = x; mi = v; }       // strictly greater -> first index
    }
    rv[tid] = mx; ri[tid] = mi;
    __syncthreads();
    for (int st = 128; st > 0; st >>= 1) {
        if (tid < st) {
            float ov = rv[tid + st]; int oi = ri[tid + st];
            if (ov > rv[tid] || (ov == rv[tid] && oi < ri[tid])) {
                rv[tid] = ov; ri[tid] = oi;
            }
        }
        __syncthreads();
    }
    float gmax = rv[0]; int gidx = ri[0];
    __syncthreads();

    float s = 0.f;
    for (int v = tid; v < OUT_DIM; v += 256) s += __expf(row[v] - gmax);
    rv[tid] = s;
    __syncthreads();
    for (int st = 128; st > 0; st >>= 1) {
        if (tid < st) rv[tid] += rv[tid + st];
        __syncthreads();
    }
    float inv = 1.f / rv[0];

    float* orow = out + ((size_t)b * MAXLEN + t) * OUT_DIM;
    for (int v = tid; v < OUT_DIM; v += 256) orow[v] = __expf(row[v] - gmax) * inv;
    if (tid == 0) y[b] = gidx;
}

// =====================================================================
// Host-side orchestration
// =====================================================================
static inline void launch_gemm(const __bf16* A, const __bf16* BT, const float* bias,
                               float* C, __bf16* Cb, int M, int N, int K, int act,
                               hipStream_t s)
{
    // NB=4 for big parallel GEMMs; NB=2 when strip count would starve the
    // device of concurrent waves (latency-bound recurrent / small GEMMs).
    long strips4 = (long)(M / 16) * (N / 64);
    if ((N % 64 == 0) && strips4 >= 512) {
        long wgs = (strips4 + 7) / 8;
        if (wgs > 4096) wgs = 4096;
        gemm_bf16_wmma<4><<<dim3((unsigned)wgs), dim3(256), 0, s>>>(
            A, BT, bias, C, Cb, M, N, K, act);
    } else {
        long strips2 = (long)(M / 16) * (N / 32);
        long wgs = (strips2 + 7) / 8;
        if (wgs > 4096) wgs = 4096;
        gemm_bf16_wmma<2><<<dim3((unsigned)wgs), dim3(256), 0, s>>>(
            A, BT, bias, C, Cb, M, N, K, act);
    }
}

extern "C" void kernel_launch(void* const* d_in, const int* in_sizes, int n_in,
                              void* d_out, int out_size, void* d_ws, size_t ws_size,
                              hipStream_t stream)
{
    // ------- inputs (setup_inputs order) -------
    const int*   source  = (const int*)  d_in[0];
    const float* enc_emb = (const float*)d_in[1];
    const float* enc_K   = (const float*)d_in[2];
    const float* enc_R   = (const float*)d_in[3];
    const float* enc_b   = (const float*)d_in[4];
    const float* dec_emb = (const float*)d_in[5];
    const float* dec_K   = (const float*)d_in[6];
    const float* dec_R   = (const float*)d_in[7];
    const float* dec_b   = (const float*)d_in[8];
    const float* W_a     = (const float*)d_in[9];
    const float* W_c     = (const float*)d_in[10];
    const float* b_c     = (const float*)d_in[11];
    const float* W_out   = (const float*)d_in[12];
    const float* b_out   = (const float*)d_in[13];
    float* out = (float*)d_out;

    // ------- workspace carve-up -------
    char* wp = (char*)d_ws;
    auto alloc = [&](size_t bytes) -> char* {
        char* p = wp;
        wp += (bytes + 255) & ~(size_t)255;
        return p;
    };
    __bf16* KR_encT = (__bf16*)alloc((size_t)4 * HID * 2 * HID * 2);   // [4096][2048]
    __bf16* KR_decT = (__bf16*)alloc((size_t)4 * HID * 2 * HID * 2);
    __bf16* WaT     = (__bf16*)alloc((size_t)HID * HID * 2);           // [1024][1024]
    __bf16* WcT     = (__bf16*)alloc((size_t)HID * 2 * HID * 2);       // [1024][2048]
    __bf16* WoutT   = (__bf16*)alloc((size_t)OUT_DIM * HID * 2);       // [32000][1024]
    __bf16* Abuf    = (__bf16*)alloc((size_t)BSZ * 2 * HID * 2);       // [64][2048]
    float*  z       = (float*) alloc((size_t)BSZ * 4 * HID * 4);       // [64][4096]
    float*  h       = (float*) alloc((size_t)BSZ * HID * 4);
    float*  c       = (float*) alloc((size_t)BSZ * HID * 4);
    float*  hs      = (float*) alloc((size_t)BSZ * SRC_LEN * HID * 4);
    __bf16* hs_bf   = (__bf16*)alloc((size_t)BSZ * SRC_LEN * HID * 2);
    __bf16* hsW_bf  = (__bf16*)alloc((size_t)BSZ * SRC_LEN * HID * 2);
    __bf16* catb    = (__bf16*)alloc((size_t)BSZ * 2 * HID * 2);
    __bf16* ah_bf   = (__bf16*)alloc((size_t)BSZ * HID * 2);
    float*  logits  = (float*) alloc((size_t)BSZ * OUT_DIM * 4);
    int*    ybuf    = (int*)   alloc((size_t)BSZ * 4);

    const dim3 tb(256);
    const dim3 ew((BSZ * HID + 255) / 256);

    // ------- one-time weight transpose + bf16 convert -------
    transpose_bf16_kernel<<<4096, tb, 0, stream>>>(enc_K, KR_encT, HID, 4 * HID, 2 * HID, 0);
    transpose_bf16_kernel<<<4096, tb, 0, stream>>>(enc_R, KR_encT, HID, 4 * HID, 2 * HID, HID);
    transpose_bf16_kernel<<<4096, tb, 0, stream>>>(dec_K, KR_decT, HID, 4 * HID, 2 * HID, 0);
    transpose_bf16_kernel<<<4096, tb, 0, stream>>>(dec_R, KR_decT, HID, 4 * HID, 2 * HID, HID);
    transpose_bf16_kernel<<<2048, tb, 0, stream>>>(W_a,   WaT,   HID,     HID,     HID,     0);
    transpose_bf16_kernel<<<4096, tb, 0, stream>>>(W_c,   WcT,   2 * HID, HID,     2 * HID, 0);
    transpose_bf16_kernel<<<8192, tb, 0, stream>>>(W_out, WoutT, HID,     OUT_DIM, HID,     0);

    // ------- encoder recurrence (256 sequential steps) -------
    enc_init_kernel<<<ew, tb, 0, stream>>>(source, enc_emb, h, c, Abuf);
    for (int t = 0; t < SRC_LEN; ++t) {
        launch_gemm(Abuf, KR_encT, enc_b, z, nullptr, BSZ, 4 * HID, 2 * HID, 0, stream);
        lstm_cell_kernel<<<ew, tb, 0, stream>>>(z, h, c, source, t, hs, hs_bf, Abuf, enc_emb);
    }

    // ------- hsW = hs @ W_a (one big parallel GEMM, bf16 output) -------
    launch_gemm(hs_bf, WaT, nullptr, nullptr, hsW_bf, BSZ * SRC_LEN, HID, HID, 0, stream);

    // ------- decoder (32 sequential steps) -------
    for (int t = 0; t < MAXLEN; ++t) {
        dec_embed_kernel<<<ew, tb, 0, stream>>>(dec_emb, ybuf, t, Abuf);
        launch_gemm(Abuf, KR_decT, dec_b, z, nullptr, BSZ, 4 * HID, 2 * HID, 0, stream);
        lstm_cell_kernel<<<ew, tb, 0, stream>>>(z, h, c, nullptr, t, nullptr, nullptr,
                                                Abuf, nullptr);
        attention_kernel<<<BSZ, tb, 0, stream>>>(h, hsW_bf, hs, source, catb);
        launch_gemm(catb, WcT, b_c, nullptr, ah_bf, BSZ, HID, 2 * HID, 1 /*tanh*/, stream);
        launch_gemm(ah_bf, WoutT, b_out, logits, nullptr, BSZ, OUT_DIM, HID, 0, stream);
        softmax_argmax_kernel<<<BSZ, tb, 0, stream>>>(logits, t, out, ybuf);
    }
    (void)in_sizes; (void)n_in; (void)out_size; (void)ws_size;
}
